// NeuroSATNet_9672266350951
// MI455X (gfx1250) — matrix operand order repaired and checked by reading, not compiled, not run on hardware
//
#include <hip/hip_runtime.h>
#include <hip/hip_bf16.h>

// ============================================================================
// NeuroSAT on MI455X (gfx1250), wave32 + v_wmma_f32_16x16x32_bf16 everywhere.
//
// Dims: B=16, V=1024, C=4096, H=128, ITERS=16.
// Strategy: bf16-convert x once (128MB -> L2-resident across all 16
// iterations), materialize x^T once for the clause-side einsum, store the
// Lmsg/Cmsg MLP outputs transposed so einsum B-fragments load contiguously,
// fuse gate-GEMM + LSTM + LayerNorm per side, and use 2x M-tile register
// blocking in all standalone GEMMs (each B fragment feeds 2 WMMAs).
// ============================================================================

#define BSZ      16
#define N_VAR    1024
#define N_CLAUSE 4096
#define HDIM     128
#define ITERS    16
#define EPSF     1e-5f

typedef __attribute__((ext_vector_type(16))) __bf16 v16bf;
typedef __attribute__((ext_vector_type(8)))  __bf16 v8bf;
typedef __attribute__((ext_vector_type(8)))  float  v8f;

// ---------------------------------------------------------------------------
// WMMA helpers (CDNA5 wave32 fragment layouts per CDNA5 ISA)
// ---------------------------------------------------------------------------

__device__ inline v8f zero_v8f() {
  v8f z;
#pragma unroll
  for (int i = 0; i < 8; i++) z[i] = 0.0f;
  return z;
}

__device__ inline v8f wmma_bf16(v16bf a, v16bf b, v8f c) {
  // 8 args: (neg_a, A, neg_b, B, c_mod, C, reuse_a, reuse_b)
  return __builtin_amdgcn_wmma_f32_16x16x32_bf16(false, a, false, b,
                                                 (short)0, c, false, false);
}

// A fragment, 16x32 bf16 from row-major A[m][k], lda in elements.
// Lane half h, m = m0 + (lane&15); element t holds k = k0 + 16h + t
// => one contiguous 16-element (32B) load per lane.
__device__ inline v16bf load_a_rm(const __bf16* __restrict__ A, int lda,
                                  int m0, int k0, int lane) {
  int half = (lane >> 4) & 1;
  int m    = m0 + (lane & 15);
  return *(const v16bf*)(A + (size_t)m * lda + k0 + half * 16);
}

// Same but with the NeuroSAT literal "flip": row index m -> m ^ 1 (pair swap).
__device__ inline v16bf load_a_rm_flip(const __bf16* __restrict__ A, int lda,
                                       int m0, int k0, int lane) {
  int half = (lane >> 4) & 1;
  int m    = (m0 + (lane & 15)) ^ 1;
  return *(const v16bf*)(A + (size_t)m * lda + k0 + half * 16);
}

// B fragment, 32x16 bf16 KxN, from B^T stored row-major [N][K] (ldk elems).
// Lane half h, n = n0 + (lane&15); element t holds k = k0 + 16h + t
// => one contiguous 16-element (32B) load per lane.
__device__ inline v16bf load_b_nk(const __bf16* __restrict__ Wt, int ldk,
                                  int n0, int k0, int lane) {
  int half = (lane >> 4) & 1;
  int n    = n0 + (lane & 15);
  return *(const v16bf*)(Wt + (size_t)n * ldk + k0 + half * 16);
}

__device__ inline float sigmoidf(float x) { return 1.0f / (1.0f + __expf(-x)); }

// ---------------------------------------------------------------------------
// Conversion / transpose / init kernels
// ---------------------------------------------------------------------------

__global__ void cvt_f32_bf16_kernel(const float* __restrict__ s,
                                    __bf16* __restrict__ d, size_t n) {
  size_t i = (size_t)blockIdx.x * blockDim.x + threadIdx.x;
  size_t stride = (size_t)gridDim.x * blockDim.x;
  for (; i < n; i += stride) d[i] = (__bf16)s[i];
}

// x [B][V][C] f32  ->  xT [B][C][V] bf16, 32x32 LDS tiles.
__global__ void xt_kernel(const float* __restrict__ x, __bf16* __restrict__ xT) {
  __shared__ float tile[32][33];
  int b  = blockIdx.z;
  int c0 = blockIdx.x * 32;
  int v0 = blockIdx.y * 32;
  for (int i = 0; i < 32; i += 8) {
    int v = v0 + threadIdx.y + i;
    int c = c0 + threadIdx.x;
    tile[threadIdx.y + i][threadIdx.x] = x[((size_t)b * N_VAR + v) * N_CLAUSE + c];
  }
  __syncthreads();
  for (int i = 0; i < 32; i += 8) {
    int c = c0 + threadIdx.y + i;
    int v = v0 + threadIdx.x;
    xT[((size_t)b * N_CLAUSE + c) * N_VAR + v] = (__bf16)tile[threadIdx.x][threadIdx.y + i];
  }
}

// Initial states: L0 = Lw+Lb bcast, L = LN(L0); C0 = Cw+Cb; Lc=Cc=0.
__global__ void init_kernel(const float* __restrict__ Lw, const float* __restrict__ Lb,
                            const float* __restrict__ Cw, const float* __restrict__ Cb,
                            const float* __restrict__ Lng, const float* __restrict__ Lnb,
                            __bf16* __restrict__ Lbf, __bf16* __restrict__ Lhbf,
                            float* __restrict__ Lc,
                            __bf16* __restrict__ Chbf, float* __restrict__ Cc) {
  int row = blockIdx.x;           // 0 .. B*N_CLAUSE-1
  int h   = threadIdx.x;          // 0 .. 127
  __shared__ float v[HDIM];
  float lv = Lw[h] + Lb[h];
  v[h] = lv;
  __syncthreads();
  float s = 0.f, ss = 0.f;
  for (int i = 0; i < HDIM; i++) { s += v[i]; ss += v[i] * v[i]; }
  float mean = s * (1.0f / HDIM);
  float var  = ss * (1.0f / HDIM) - mean * mean;
  float ln   = (lv - mean) * rsqrtf(var + EPSF) * Lng[h] + Lnb[h];
  float cv   = Cw[h] + Cb[h];
  size_t ci  = (size_t)row * HDIM + h;
  Chbf[ci] = (__bf16)cv;
  Cc[ci]   = 0.0f;
  if (row < BSZ * N_VAR) {
    Lbf[ci]  = (__bf16)ln;
    Lhbf[ci] = (__bf16)lv;
    Lc[ci]   = 0.0f;
  }
}

// ---------------------------------------------------------------------------
// Generic WMMA GEMM: D = act(A[M,K] @ Wt[N,K]^T + bias), N == 128.
// Block = 256 threads = 8 waves; each wave: 2 M-tiles (32 rows) x 128 cols.
// Each B fragment load feeds 2 WMMAs (register blocking).
// TROUT: store D transposed as [N][M] (used for Lmsg/Cmsg so that the einsum
// GEMMs get contiguous B-fragment loads).
// ---------------------------------------------------------------------------
template <int ACT, int TROUT>
__global__ __launch_bounds__(256)
void gemm_kernel(const __bf16* __restrict__ A, const __bf16* __restrict__ Wt,
                 const float* __restrict__ bias, __bf16* __restrict__ D,
                 int M, int K) {
  int lane = threadIdx.x & 31;
  int wave = threadIdx.x >> 5;
  int m0   = (blockIdx.x * 8 + wave) * 32;
  if (m0 >= M) return;

  v8f acc[2][8];
#pragma unroll
  for (int mt = 0; mt < 2; mt++)
#pragma unroll
    for (int t = 0; t < 8; t++) acc[mt][t] = zero_v8f();

  for (int k0 = 0; k0 < K; k0 += 32) {
    v16bf a0 = load_a_rm(A, K, m0, k0, lane);
    v16bf a1 = load_a_rm(A, K, m0 + 16, k0, lane);
#pragma unroll
    for (int nt = 0; nt < 8; nt++) {
      v16bf b = load_b_nk(Wt, K, nt * 16, k0, lane);
      acc[0][nt] = wmma_bf16(a0, b, acc[0][nt]);
      acc[1][nt] = wmma_bf16(a1, b, acc[1][nt]);
    }
  }

  int half = (lane >> 4) & 1;
  int ln   = lane & 15;
#pragma unroll
  for (int mt = 0; mt < 2; mt++) {
    int mbase = m0 + mt * 16;
#pragma unroll
    for (int nt = 0; nt < 8; nt++) {
      int n = nt * 16 + ln;
      float bv = bias ? bias[n] : 0.0f;
      if (TROUT) {
        __bf16* p = D + (size_t)n * M + mbase + 8 * half;
#pragma unroll
        for (int r = 0; r < 8; r++) {
          float v = acc[mt][nt][r] + bv;
          if (ACT) v = fmaxf(v, 0.0f);
          p[r] = (__bf16)v;
        }
      } else {
#pragma unroll
        for (int r = 0; r < 8; r++) {
          float v = acc[mt][nt][r] + bv;
          if (ACT) v = fmaxf(v, 0.0f);
          D[(size_t)(mbase + r + 8 * half) * HDIM + n] = (__bf16)v;
        }
      }
    }
  }
}

// ---------------------------------------------------------------------------
// Batched einsum GEMM: D[b] = A[b][M,K] @ Bt[b][128,K]^T.  (N == 128, no bias)
// Used for  ML = x^T @ Lmsg   (A = xT,  Bt = LmsgT)
// and       MC = x   @ Cmsg   (A = xb,  Bt = CmsgT)
// Same 2x M-tile register blocking as gemm_kernel.
// ---------------------------------------------------------------------------
__global__ __launch_bounds__(256)
void gemm_x_kernel(const __bf16* __restrict__ A, size_t aStride, int lda,
                   const __bf16* __restrict__ Bt, size_t bOff, int ldk,
                   __bf16* __restrict__ D, size_t dStride, int M, int K) {
  int b    = blockIdx.y;
  int lane = threadIdx.x & 31;
  int wave = threadIdx.x >> 5;
  int m0   = (blockIdx.x * 8 + wave) * 32;
  if (m0 >= M) return;

  const __bf16* Ab = A + (size_t)b * aStride;
  const __bf16* Bb = Bt + (size_t)b * bOff;   // column offset into [128, B*rows]
  __bf16*       Db = D + (size_t)b * dStride;

  v8f acc[2][8];
#pragma unroll
  for (int mt = 0; mt < 2; mt++)
#pragma unroll
    for (int t = 0; t < 8; t++) acc[mt][t] = zero_v8f();

  for (int k0 = 0; k0 < K; k0 += 32) {
    v16bf a0 = load_a_rm(Ab, lda, m0, k0, lane);
    v16bf a1 = load_a_rm(Ab, lda, m0 + 16, k0, lane);
#pragma unroll
    for (int nt = 0; nt < 8; nt++) {
      v16bf bb = load_b_nk(Bb, ldk, nt * 16, k0, lane);
      acc[0][nt] = wmma_bf16(a0, bb, acc[0][nt]);
      acc[1][nt] = wmma_bf16(a1, bb, acc[1][nt]);
    }
  }

  int half = (lane >> 4) & 1;
  int ln   = lane & 15;
#pragma unroll
  for (int mt = 0; mt < 2; mt++) {
    int mbase = m0 + mt * 16;
#pragma unroll
    for (int nt = 0; nt < 8; nt++) {
      int n = nt * 16 + ln;
#pragma unroll
      for (int r = 0; r < 8; r++)
        Db[(size_t)(mbase + r + 8 * half) * HDIM + n] = (__bf16)acc[mt][nt][r];
    }
  }
}

// ---------------------------------------------------------------------------
// Fused clause LSTM: gates = ML@Wih^T + Ch@Whh^T + b;  Ch,Cc update; C = LN(Ch).
// Block = 512 threads = 16 waves, owns 16 rows; gates staged in 32KB LDS.
// ---------------------------------------------------------------------------
__global__ __launch_bounds__(512)
void lstm_c_kernel(const __bf16* __restrict__ ML, __bf16* __restrict__ Chbf,
                   float* __restrict__ Cc,
                   const __bf16* __restrict__ Wih, const __bf16* __restrict__ Whh,
                   const float* __restrict__ bih, const float* __restrict__ bhh,
                   const float* __restrict__ ng, const float* __restrict__ nb,
                   __bf16* __restrict__ Cnbf) {
  __shared__ float gates[16][4 * HDIM];
  int lane = threadIdx.x & 31;
  int wave = threadIdx.x >> 5;
  int r0   = blockIdx.x * 16;
  int n0   = wave * 32;

  v8f acc0 = zero_v8f(), acc1 = zero_v8f();
  for (int k0 = 0; k0 < HDIM; k0 += 32) {          // input GEMM: ML @ Wih^T
    v16bf a  = load_a_rm(ML, HDIM, r0, k0, lane);
    v16bf b0 = load_b_nk(Wih, HDIM, n0, k0, lane);
    v16bf b1 = load_b_nk(Wih, HDIM, n0 + 16, k0, lane);
    acc0 = wmma_bf16(a, b0, acc0);
    acc1 = wmma_bf16(a, b1, acc1);
  }
  for (int k0 = 0; k0 < HDIM; k0 += 32) {          // hidden GEMM: Ch @ Whh^T
    v16bf a  = load_a_rm(Chbf, HDIM, r0, k0, lane);
    v16bf b0 = load_b_nk(Whh, HDIM, n0, k0, lane);
    v16bf b1 = load_b_nk(Whh, HDIM, n0 + 16, k0, lane);
    acc0 = wmma_bf16(a, b0, acc0);
    acc1 = wmma_bf16(a, b1, acc1);
  }

  int half = (lane >> 4) & 1;
  int ln   = lane & 15;
  {
    int n = n0 + ln;
    float bs = bih[n] + bhh[n];
#pragma unroll
    for (int r = 0; r < 8; r++) gates[r + 8 * half][n] = acc0[r] + bs;
    n = n0 + 16 + ln;
    bs = bih[n] + bhh[n];
#pragma unroll
    for (int r = 0; r < 8; r++) gates[r + 8 * half][n] = acc1[r] + bs;
  }
  __syncthreads();

  // Elementwise LSTM + LayerNorm: wave w owns row w; lane owns 4 h-slots.
  int row = r0 + wave;
  float h2v[4];
  float s = 0.f, ss = 0.f;
#pragma unroll
  for (int j = 0; j < 4; j++) {
    int h = lane * 4 + j;
    float ig = gates[wave][h];
    float fg = gates[wave][HDIM + h];
    float gg = gates[wave][2 * HDIM + h];
    float og = gates[wave][3 * HDIM + h];
    float c  = Cc[(size_t)row * HDIM + h];
    float c2 = sigmoidf(fg) * c + sigmoidf(ig) * tanhf(gg);
    float h2 = sigmoidf(og) * tanhf(c2);
    Cc[(size_t)row * HDIM + h] = c2;
    h2v[j] = h2;
    s += h2;
    ss += h2 * h2;
  }
#pragma unroll
  for (int off = 16; off; off >>= 1) {
    s  += __shfl_xor(s, off, 32);
    ss += __shfl_xor(ss, off, 32);
  }
  float mean = s * (1.0f / HDIM);
  float var  = ss * (1.0f / HDIM) - mean * mean;
  float rstd = rsqrtf(var + EPSF);
#pragma unroll
  for (int j = 0; j < 4; j++) {
    int h = lane * 4 + j;
    float hn = (h2v[j] - mean) * rstd * ng[h] + nb[h];
    Chbf[(size_t)row * HDIM + h] = (__bf16)h2v[j];
    Cnbf[(size_t)row * HDIM + h] = (__bf16)hn;
  }
}

// ---------------------------------------------------------------------------
// Fused literal LSTM: gates = [MC, flip(L)]@Wih^T + Lh@Whh^T + b;
// Lh,Lc update; L = LN(Lh).   Wih is [512,256].
// ---------------------------------------------------------------------------
__global__ __launch_bounds__(512)
void lstm_l_kernel(const __bf16* __restrict__ MC, __bf16* __restrict__ Lbf,
                   __bf16* __restrict__ Lhbf, float* __restrict__ Lc,
                   const __bf16* __restrict__ Wih, const __bf16* __restrict__ Whh,
                   const float* __restrict__ bih, const float* __restrict__ bhh,
                   const float* __restrict__ ng, const float* __restrict__ nb) {
  __shared__ float gates[16][4 * HDIM];
  int lane = threadIdx.x & 31;
  int wave = threadIdx.x >> 5;
  int r0   = blockIdx.x * 16;
  int n0   = wave * 32;

  v8f acc0 = zero_v8f(), acc1 = zero_v8f();
  for (int k0 = 0; k0 < HDIM; k0 += 32) {          // MC @ Wih[:, 0:128]^T
    v16bf a  = load_a_rm(MC, HDIM, r0, k0, lane);
    v16bf b0 = load_b_nk(Wih, 2 * HDIM, n0, k0, lane);
    v16bf b1 = load_b_nk(Wih, 2 * HDIM, n0 + 16, k0, lane);
    acc0 = wmma_bf16(a, b0, acc0);
    acc1 = wmma_bf16(a, b1, acc1);
  }
  for (int k0 = 0; k0 < HDIM; k0 += 32) {          // flip(L) @ Wih[:,128:256]^T
    v16bf a  = load_a_rm_flip(Lbf, HDIM, r0, k0, lane);
    v16bf b0 = load_b_nk(Wih, 2 * HDIM, n0, HDIM + k0, lane);
    v16bf b1 = load_b_nk(Wih, 2 * HDIM, n0 + 16, HDIM + k0, lane);
    acc0 = wmma_bf16(a, b0, acc0);
    acc1 = wmma_bf16(a, b1, acc1);
  }
  for (int k0 = 0; k0 < HDIM; k0 += 32) {          // Lh @ Whh^T
    v16bf a  = load_a_rm(Lhbf, HDIM, r0, k0, lane);
    v16bf b0 = load_b_nk(Whh, HDIM, n0, k0, lane);
    v16bf b1 = load_b_nk(Whh, HDIM, n0 + 16, k0, lane);
    acc0 = wmma_bf16(a, b0, acc0);
    acc1 = wmma_bf16(a, b1, acc1);
  }

  int half = (lane >> 4) & 1;
  int ln   = lane & 15;
  {
    int n = n0 + ln;
    float bs = bih[n] + bhh[n];
#pragma unroll
    for (int r = 0; r < 8; r++) gates[r + 8 * half][n] = acc0[r] + bs;
    n = n0 + 16 + ln;
    bs = bih[n] + bhh[n];
#pragma unroll
    for (int r = 0; r < 8; r++) gates[r + 8 * half][n] = acc1[r] + bs;
  }
  __syncthreads();

  int row = r0 + wave;
  float h2v[4];
  float s = 0.f, ss = 0.f;
#pragma unroll
  for (int j = 0; j < 4; j++) {
    int h = lane * 4 + j;
    float ig = gates[wave][h];
    float fg = gates[wave][HDIM + h];
    float gg = gates[wave][2 * HDIM + h];
    float og = gates[wave][3 * HDIM + h];
    float c  = Lc[(size_t)row * HDIM + h];
    float c2 = sigmoidf(fg) * c + sigmoidf(ig) * tanhf(gg);
    float h2 = sigmoidf(og) * tanhf(c2);
    Lc[(size_t)row * HDIM + h] = c2;
    h2v[j] = h2;
    s += h2;
    ss += h2 * h2;
  }
#pragma unroll
  for (int off = 16; off; off >>= 1) {
    s  += __shfl_xor(s, off, 32);
    ss += __shfl_xor(ss, off, 32);
  }
  float mean = s * (1.0f / HDIM);
  float var  = ss * (1.0f / HDIM) - mean * mean;
  float rstd = rsqrtf(var + EPSF);
#pragma unroll
  for (int j = 0; j < 4; j++) {
    int h = lane * 4 + j;
    float hn = (h2v[j] - mean) * rstd * ng[h] + nb[h];
    Lhbf[(size_t)row * HDIM + h] = (__bf16)h2v[j];
    Lbf[(size_t)row * HDIM + h]  = (__bf16)hn;
  }
}

// ---------------------------------------------------------------------------
// Final vote reduction: out[b] = sigmoid(sum_v mask * (t2[b,v,:].w3 + b3))
// ---------------------------------------------------------------------------
__global__ __launch_bounds__(256)
void vote_kernel(const __bf16* __restrict__ T, const float* __restrict__ w3,
                 const float* __restrict__ b3, const float* __restrict__ mask,
                 float* __restrict__ out) {
  int b = blockIdx.x;
  __shared__ float red[256];
  float acc = 0.f;
  for (int v = threadIdx.x; v < N_VAR; v += 256) {
    const __bf16* row = T + ((size_t)b * N_VAR + v) * HDIM;
    float d = 0.f;
    for (int h = 0; h < HDIM; h++) d += (float)row[h] * w3[h];
    acc += (d + b3[0]) * mask[b * N_VAR + v];
  }
  red[threadIdx.x] = acc;
  __syncthreads();
  for (int s2 = 128; s2 > 0; s2 >>= 1) {
    if ((int)threadIdx.x < s2) red[threadIdx.x] += red[threadIdx.x + s2];
    __syncthreads();
  }
  if (threadIdx.x == 0) out[b] = 1.0f / (1.0f + __expf(-red[0]));
}

// ===========================================================================
// Host launcher
// ===========================================================================
extern "C" void kernel_launch(void* const* d_in, const int* in_sizes, int n_in,
                              void* d_out, int out_size, void* d_ws, size_t ws_size,
                              hipStream_t stream) {
  const float* x        = (const float*)d_in[0];
  const float* mask     = (const float*)d_in[1];
  const float* L_init_w = (const float*)d_in[2];
  const float* L_init_b = (const float*)d_in[3];
  const float* C_init_w = (const float*)d_in[4];
  const float* C_init_b = (const float*)d_in[5];
  const float* Lmsg_w[3] = {(const float*)d_in[6],  (const float*)d_in[8],  (const float*)d_in[10]};
  const float* Lmsg_b[3] = {(const float*)d_in[7],  (const float*)d_in[9],  (const float*)d_in[11]};
  const float* Cmsg_w[3] = {(const float*)d_in[12], (const float*)d_in[14], (const float*)d_in[16]};
  const float* Cmsg_b[3] = {(const float*)d_in[13], (const float*)d_in[15], (const float*)d_in[17]};
  const float* Lup_wih = (const float*)d_in[18];
  const float* Lup_whh = (const float*)d_in[19];
  const float* Lup_bih = (const float*)d_in[20];
  const float* Lup_bhh = (const float*)d_in[21];
  const float* Cup_wih = (const float*)d_in[22];
  const float* Cup_whh = (const float*)d_in[23];
  const float* Cup_bih = (const float*)d_in[24];
  const float* Cup_bhh = (const float*)d_in[25];
  const float* Lnorm_g = (const float*)d_in[26];
  const float* Lnorm_b = (const float*)d_in[27];
  const float* Cnorm_g = (const float*)d_in[28];
  const float* Cnorm_b = (const float*)d_in[29];
  const float* Lvote_w[2] = {(const float*)d_in[30], (const float*)d_in[32]};
  const float* Lvote_b[2] = {(const float*)d_in[31], (const float*)d_in[33]};
  const float* Lvote_w3 = (const float*)d_in[34];
  const float* Lvote_b3 = (const float*)d_in[35];
  float* out = (float*)d_out;

  const size_t NX   = (size_t)BSZ * N_VAR * N_CLAUSE;     // 67,108,864
  const size_t LROW = (size_t)BSZ * N_VAR;                // 16384
  const size_t CROW = (size_t)BSZ * N_CLAUSE;             // 65536

  // ---- carve workspace (~429 MB total) ----
  char* w = (char*)d_ws;
  size_t off = 0;
  auto carve = [&](size_t bytes) -> void* {
    void* p = w + off;
    off += (bytes + 255) & ~(size_t)255;
    return p;
  };
  __bf16* xb  = (__bf16*)carve(NX * 2);                    // x  bf16 [B][V][C]
  __bf16* xT  = (__bf16*)carve(NX * 2);                    // xT bf16 [B][C][V]
  __bf16* wLmsg[3], *wCmsg[3], *wLvote[2];
  for (int i = 0; i < 3; i++) wLmsg[i] = (__bf16*)carve(HDIM * HDIM * 2);
  for (int i = 0; i < 3; i++) wCmsg[i] = (__bf16*)carve(HDIM * HDIM * 2);
  for (int i = 0; i < 2; i++) wLvote[i] = (__bf16*)carve(HDIM * HDIM * 2);
  __bf16* wLih = (__bf16*)carve((size_t)512 * 256 * 2);
  __bf16* wLhh = (__bf16*)carve((size_t)512 * 128 * 2);
  __bf16* wCih = (__bf16*)carve((size_t)512 * 128 * 2);
  __bf16* wChh = (__bf16*)carve((size_t)512 * 128 * 2);
  __bf16* Lbf   = (__bf16*)carve(LROW * HDIM * 2);         // L (normed) state
  __bf16* Lhbf  = (__bf16*)carve(LROW * HDIM * 2);         // Lh state
  float*  Lc    = (float*) carve(LROW * HDIM * 4);         // Lc state (f32)
  __bf16* Chbf  = (__bf16*)carve(CROW * HDIM * 2);         // Ch state
  __bf16* Cnbf  = (__bf16*)carve(CROW * HDIM * 2);         // C = LN(Ch)
  float*  Cc    = (float*) carve(CROW * HDIM * 4);         // Cc state (f32)
  __bf16* LmsgT = (__bf16*)carve(LROW * HDIM * 2);         // [128][B*V] transposed
  __bf16* CmsgT = (__bf16*)carve(CROW * HDIM * 2);         // [128][B*C] transposed
  __bf16* MLb   = (__bf16*)carve(CROW * HDIM * 2);
  __bf16* MCb   = (__bf16*)carve(LROW * HDIM * 2);
  __bf16* t0    = (__bf16*)carve(CROW * HDIM * 2);
  __bf16* t1    = (__bf16*)carve(CROW * HDIM * 2);

  // ---- one-time prep: bf16 x (fits 192MB L2), x^T, bf16 weights ----
  cvt_f32_bf16_kernel<<<4096, 256, 0, stream>>>(x, xb, NX);
  xt_kernel<<<dim3(N_CLAUSE / 32, N_VAR / 32, BSZ), dim3(32, 8), 0, stream>>>(x, xT);
  auto cvt = [&](const float* s, __bf16* d, size_t n) {
    int grid = (int)((n + 255) / 256);
    if (grid > 1024) grid = 1024;
    cvt_f32_bf16_kernel<<<grid, 256, 0, stream>>>(s, d, n);
  };
  for (int i = 0; i < 3; i++) cvt(Lmsg_w[i], wLmsg[i], HDIM * HDIM);
  for (int i = 0; i < 3; i++) cvt(Cmsg_w[i], wCmsg[i], HDIM * HDIM);
  for (int i = 0; i < 2; i++) cvt(Lvote_w[i], wLvote[i], HDIM * HDIM);
  cvt(Lup_wih, wLih, (size_t)512 * 256);
  cvt(Lup_whh, wLhh, (size_t)512 * 128);
  cvt(Cup_wih, wCih, (size_t)512 * 128);
  cvt(Cup_whh, wChh, (size_t)512 * 128);

  init_kernel<<<(int)CROW, HDIM, 0, stream>>>(L_init_w, L_init_b, C_init_w, C_init_b,
                                              Lnorm_g, Lnorm_b,
                                              Lbf, Lhbf, Lc, Chbf, Cc);

  // ---- 16 message-passing iterations ----
  for (int it = 0; it < ITERS; it++) {
    // Lmsg = MLP(L); last layer stored transposed [128][B*V]
    gemm_kernel<1, 0><<<(int)(LROW / 256), 256, 0, stream>>>(Lbf, wLmsg[0], Lmsg_b[0], t0, (int)LROW, HDIM);
    gemm_kernel<1, 0><<<(int)(LROW / 256), 256, 0, stream>>>(t0, wLmsg[1], Lmsg_b[1], t1, (int)LROW, HDIM);
    gemm_kernel<0, 1><<<(int)(LROW / 256), 256, 0, stream>>>(t1, wLmsg[2], Lmsg_b[2], LmsgT, (int)LROW, HDIM);
    // ML[b] = x[b]^T @ Lmsg[b]   (A = xT, lda=V; Bt = LmsgT + b*V, ldk=B*V)
    gemm_x_kernel<<<dim3(N_CLAUSE / 256, BSZ), 256, 0, stream>>>(
        xT, (size_t)N_CLAUSE * N_VAR, N_VAR,
        LmsgT, (size_t)N_VAR, (int)LROW,
        MLb, (size_t)N_CLAUSE * HDIM, N_CLAUSE, N_VAR);
    // clause LSTM + LayerNorm (fused)
    lstm_c_kernel<<<(int)(CROW / 16), 512, 0, stream>>>(MLb, Chbf, Cc, wCih, wChh,
                                                        Cup_bih, Cup_bhh,
                                                        Cnorm_g, Cnorm_b, Cnbf);
    // Cmsg = MLP(C); last layer transposed [128][B*C]
    gemm_kernel<1, 0><<<(int)(CROW / 256), 256, 0, stream>>>(Cnbf, wCmsg[0], Cmsg_b[0], t0, (int)CROW, HDIM);
    gemm_kernel<1, 0><<<(int)(CROW / 256), 256, 0, stream>>>(t0, wCmsg[1], Cmsg_b[1], t1, (int)CROW, HDIM);
    gemm_kernel<0, 1><<<(int)(CROW / 256), 256, 0, stream>>>(t1, wCmsg[2], Cmsg_b[2], CmsgT, (int)CROW, HDIM);
    // MC[b] = x[b] @ Cmsg[b]   (A = xb, lda=C; Bt = CmsgT + b*C, ldk=B*C)
    gemm_x_kernel<<<dim3(N_VAR / 256, BSZ), 256, 0, stream>>>(
        xb, (size_t)N_VAR * N_CLAUSE, N_CLAUSE,
        CmsgT, (size_t)N_CLAUSE, (int)CROW,
        MCb, (size_t)N_VAR * HDIM, N_VAR, N_CLAUSE);
    // literal LSTM + LayerNorm (fused, includes flip(L) concat)
    lstm_l_kernel<<<(int)(LROW / 16), 512, 0, stream>>>(MCb, Lbf, Lhbf, Lc,
                                                        wLih, wLhh,
                                                        Lup_bih, Lup_bhh,
                                                        Lnorm_g, Lnorm_b);
  }

  // ---- vote MLP + masked sum + sigmoid ----
  gemm_kernel<1, 0><<<(int)(LROW / 256), 256, 0, stream>>>(Lbf, wLvote[0], Lvote_b[0], t0, (int)LROW, HDIM);
  gemm_kernel<1, 0><<<(int)(LROW / 256), 256, 0, stream>>>(t0, wLvote[1], Lvote_b[1], t1, (int)LROW, HDIM);
  vote_kernel<<<BSZ, 256, 0, stream>>>(t1, Lvote_w3, Lvote_b3, mask, out);
}
